// Dynamics_10892037063055
// MI455X (gfx1250) — matrix operand order repaired and checked by reading, not compiled, not run on hardware
//
#include <hip/hip_runtime.h>
#include <math.h>

// ---------------------------------------------------------------------------
// L2HMC dynamics on gfx1250 (MI455X).
// One wave32 = one 16-row batch tile; pos/mom resident in VGPRs in WMMA C
// layout for the whole 25-step serial scan. All MLP matmuls are
// v_wmma_f32_16x16x32_f16 with weights pre-swizzled into B-fragment layout in
// LDS (shared by the 4 waves of a block). grad(potential)=x fuses Wv+Wx for
// the momentum net. Staging is skipped when the input tile is unchanged.
// ---------------------------------------------------------------------------

typedef _Float16 half_t;
typedef __attribute__((ext_vector_type(16))) _Float16 v16h;
typedef __attribute__((ext_vector_type(8)))  _Float16 v8h;
typedef __attribute__((ext_vector_type(8)))  float    v8f;

#define BATCH 4096
#define DDIM  128
#define NSTEP 25
#define NH    10
#define WPB   4                      // waves per block
#define BD    (BATCH*DDIM)

// fragment ids (each frag: 32 lanes x 16 halfs = 1 KB)
#define F_WVX_M 0                    // mom net: (Wv+Wx) K-chunks 0..3
#define F_WH_M  4
#define F_S_M   5                    // scale       (8 column blocks)
#define F_T_M   13                   // translation
#define F_Q_M   21                   // transformation
#define F_WV_P  29                   // pos net
#define F_WX_P  33
#define F_WH_P  37
#define F_S_P   38
#define F_T_P   46
#define F_Q_P   54
#define F_COUNT 62

// FA float offsets: small per-net vectors (zero-padded to 16)
#define O_HB0 0                      // bv+bx+bt  [2][16]
#define O_WT0 32                     // Wt row0   [2][16]
#define O_WT1 64                     // Wt row1   [2][16]
#define O_BH  96                     // bh        [2][16]
#define O_PK  128                    // packed {bs,btr,bq,e^cs,e^cq,pad3} [2][128][8]
#define FA_FLOATS (128 + 2*128*8)

#define LDS_FR_BYTES   (F_COUNT*512*2)
#define LDS_FA_BYTES   (FA_FLOATS*4)
#define LDS_WAVE_HALFS 7168          // tileP 2048 + tileA 2048 + tileB 2048 + hbuf 512 + h2buf 512
#define LDS_BYTES      (LDS_FR_BYTES + LDS_FA_BYTES + WPB*LDS_WAVE_HALFS*2)

// wave-private LDS ordering: CDNA5 LDS ops from one wave complete in order,
// so a compiler-only memory barrier is sufficient (no s_barrier needed).
#define WAVE_LDS_FENCE() asm volatile("" ::: "memory")

struct NetP {
  const float *cs, *ct;              // coeff_scale, coeff_transformation (1x128)
  const float *hw, *hb;              // h:     10x10, 10
  const float *sw, *sb;              // scale: 10x128, 128
  const float *tw, *tb;              // t:     2x10, 10
  const float *qw, *qb;              // transformation: 10x128, 128
  const float *trw, *trb;            // translation:    10x128, 128
  const float *vw, *vb;              // v: 128x10, 10
  const float *xw, *xb;              // x: 128x10, 10
};

// --------------------------- small math helpers ----------------------------

__device__ __forceinline__ float fast_tanh(float x) {
  x = fminf(fmaxf(x, -15.f), 15.f);
  const float e = __expf(2.0f * x);
  return (e - 1.0f) * __builtin_amdgcn_rcpf(e + 1.0f);
}

__device__ __forceinline__ v8f wmma_f16(v16h a, v16h b, v8f c) {
  return __builtin_amdgcn_wmma_f32_16x16x32_f16(false, a, false, b, (short)0, c,
                                                false, false);
}

// B-fragment (32x16 f16): lane L holds col (L&15), K=(L>>4)*8+{0..7,16..23},
// 32 contiguous bytes per lane.
__device__ __forceinline__ v16h load_frag(const half_t* __restrict__ FR, int f) {
  const int lane = threadIdx.x & 31;
  return *(const v16h*)(FR + f*512 + lane*16);
}

// A-fragment (16x32 f16) from row-major f16 tile.
__device__ __forceinline__ v16h load_a(const half_t* __restrict__ tile,
                                       int k0, int stride) {
  const int lane = threadIdx.x & 31;
  const int row  = lane & 15;
  const int kb   = k0 + ((lane >> 4) << 3);
  v16h r;
  *(v8h*)&r         = *(const v8h*)(tile + row*stride + kb);
  *(((v8h*)&r) + 1) = *(const v8h*)(tile + row*stride + kb + 16);
  return r;
}

// build one B-fragment from row-major f32 weight [kRows][nTot] (optionally
// summed with W2), zero-padding K and N out of range
__device__ __forceinline__ void build_frag(half_t* __restrict__ FR, int f,
                                           const float* W, const float* W2,
                                           int kRows, int nTot, int k0, int n0) {
  const int lane = threadIdx.x & 31;
  const int n  = n0 + (lane & 15);
  const int kb = k0 + ((lane >> 4) << 3);
  half_t* dst = FR + f*512 + lane*16;
  for (int j = 0; j < 16; ++j) {
    int k = kb + (j & 7) + ((j >> 3) << 4);
    float v = 0.f;
    if (k < kRows && n < nTot) {
      v = W[k*nTot + n];
      if (W2) v += W2[k*nTot + n];
    }
    dst[j] = (half_t)v;
  }
}

// ------------------------------ sub-step -----------------------------------
// KIND : 0 = momentum update (input = pos via tileP, mom net, fused Wvx)
//        1 = position update (inputs = mom (tileA) and masked pos (tileB))
// FWD  : 1 = forward dynamics, 0 = backward
// STAGE: 1 = (re)stage the primary input tile (it changed since last staged)
template<int KIND, int FWD, int STAGE>
__device__ __forceinline__
void substep(float (&P)[8][8], float (&M)[8][8], float (&ld)[8],
             half_t* __restrict__ tileP, half_t* __restrict__ tileA,
             half_t* __restrict__ tileB,
             half_t* __restrict__ hbuf,  half_t* __restrict__ h2buf,
             const half_t* __restrict__ FR, const float* __restrict__ FA,
             const float (&m8)[8], int mflip,
             float eps, float hbias, float b2) {
  const int lane = threadIdx.x & 31;
  const int n    = lane & 15;
  const int hg   = lane >> 4;

  // ---- stage f16 input tiles (row-major 16x128), branchless ----
  float mval[8];
  if (KIND == 0) {
    if (STAGE) {
#pragma unroll
      for (int b = 0; b < 8; ++b)
#pragma unroll
        for (int i = 0; i < 8; ++i)
          tileP[(i + 8*hg)*DDIM + 16*b + n] = (half_t)P[b][i];
    }
  } else {
#pragma unroll
    for (int b = 0; b < 8; ++b) {
      const float mv = mflip ? (1.0f - m8[b]) : m8[b];
      mval[b] = mv;
      const float mx = FWD ? mv : (1.0f - mv);   // multiplier on x-input
#pragma unroll
      for (int i = 0; i < 8; ++i) {
        const int off = (i + 8*hg)*DDIM + 16*b + n;
        if (STAGE) tileA[off] = (half_t)M[b][i];
        tileB[off] = (half_t)(mx * P[b][i]);
      }
    }
  }
  WAVE_LDS_FENCE();

  // ---- hidden layer: 16x128 @ 128x10(pad16) ----
  v8f acc = {};
  if (KIND == 0) {
#pragma unroll
    for (int c = 0; c < 4; ++c)
      acc = wmma_f16(load_a(tileP, 32*c, DDIM), load_frag(FR, F_WVX_M + c), acc);
  } else {
#pragma unroll
    for (int c = 0; c < 4; ++c)
      acc = wmma_f16(load_a(tileA, 32*c, DDIM), load_frag(FR, F_WV_P + c), acc);
#pragma unroll
    for (int c = 0; c < 4; ++c)
      acc = wmma_f16(load_a(tileB, 32*c, DDIM), load_frag(FR, F_WX_P + c), acc);
  }
#pragma unroll
  for (int i = 0; i < 8; ++i)    // pad cols give acc=0, hbias=0 -> relu 0
    hbuf[(i + 8*hg)*32 + n] = (half_t)fmaxf(acc[i] + hbias, 0.f);
  WAVE_LDS_FENCE();

  // ---- layer 2: 16x10(pad32) @ 10x10(pad16), single WMMA ----
  v8f acc2 = {};
  acc2 = wmma_f16(load_a(hbuf, 0, 32), load_frag(FR, KIND ? F_WH_P : F_WH_M), acc2);
#pragma unroll
  for (int i = 0; i < 8; ++i)
    h2buf[(i + 8*hg)*32 + n] = (half_t)fmaxf(acc2[i] + b2, 0.f);
  WAVE_LDS_FENCE();

  // ---- output layers + state update ----
  const v16h hf = load_a(h2buf, 0, 32);
  const int FS = KIND ? F_S_P : F_S_M;
  const int FT = KIND ? F_T_P : F_T_M;
  const int FQ = KIND ? F_Q_P : F_Q_M;
  const int net = KIND;

#pragma unroll
  for (int b = 0; b < 8; ++b) {
    v8f z = {};
    v8f aS = wmma_f16(hf, load_frag(FR, FS + b), z);
    v8f aT = wmma_f16(hf, load_frag(FR, FT + b), z);
    v8f aQ = wmma_f16(hf, load_frag(FR, FQ + b), z);
    const int col = 16*b + n;
    const float4 pk = *(const float4*)(FA + O_PK + net*1024 + col*8); // bs,btr,bq,e^cs
    const float  cq = FA[O_PK + net*1024 + col*8 + 4];                // e^cq
#pragma unroll
    for (int i = 0; i < 8; ++i) {
      const float sR = fast_tanh(aS[i] + pk.x) * pk.w;
      const float tr = aT[i] + pk.y;
      const float qR = fast_tanh(aQ[i] + pk.z) * cq;
      if (KIND == 0) {
        const float sE = 0.5f*eps*sR;
        const float eq = __expf(eps*qR);
        if (FWD) {
          M[b][i] = fmaf(M[b][i], __expf(sE), -0.5f*eps*fmaf(eq, P[b][i], -tr));
          ld[i] += sE;
        } else {
          M[b][i] = __expf(-sE) * fmaf(0.5f*eps, fmaf(eq, P[b][i], -tr), M[b][i]);
          ld[i] -= sE;
        }
      } else {
        const float sE = eps*sR;
        const float eq = __expf(eps*qR);
        if (FWD) {
          const float u = 1.0f - mval[b];
          const float a = fmaf(u, __expf(sE), 1.0f - u);
          P[b][i] = fmaf(P[b][i], a, (u*eps)*fmaf(eq, M[b][i], tr));
          ld[i] += u*sE;
        } else {
          const float u  = mval[b];
          const float em = __expf(-sE);
          const float a  = fmaf(u, em, 1.0f - u);
          P[b][i] = fmaf(P[b][i], a, (u*em)*fmaf(-eps*eq, M[b][i], tr));
          ld[i] -= u*sE;
        }
      }
    }
  }
}

// ------------------------------ main kernel --------------------------------

__global__ __launch_bounds__(WPB*32)
void dyn_kernel(const float* __restrict__ position, NetP posP, NetP momP,
                const float* __restrict__ epsPtr, const float* __restrict__ masks,
                const float* __restrict__ noise_f, const float* __restrict__ noise_b,
                float* __restrict__ ws) {
  extern __shared__ __align__(32) char smem[];
  half_t* FR = (half_t*)smem;
  float*  FA = (float*)(smem + LDS_FR_BYTES);
  half_t* WB = (half_t*)(smem + LDS_FR_BYTES + LDS_FA_BYTES);

  const int tid  = threadIdx.x;
  const int lane = tid & 31;
  const int wave = tid >> 5;
  const int n    = lane & 15;
  const int hg   = lane >> 4;

  // ---- one-time weight preparation (block-cooperative) ----
  for (int f = wave; f < F_COUNT; f += WPB) {
    if      (f < 4)   build_frag(FR, f, momP.vw,  momP.xw, 128, NH,  32*f,          0);
    else if (f == 4)  build_frag(FR, f, momP.hw,  nullptr, NH,  NH,  0,             0);
    else if (f < 13)  build_frag(FR, f, momP.sw,  nullptr, NH,  128, 0, 16*(f-F_S_M));
    else if (f < 21)  build_frag(FR, f, momP.trw, nullptr, NH,  128, 0, 16*(f-F_T_M));
    else if (f < 29)  build_frag(FR, f, momP.qw,  nullptr, NH,  128, 0, 16*(f-F_Q_M));
    else if (f < 33)  build_frag(FR, f, posP.vw,  nullptr, 128, NH,  32*(f-F_WV_P), 0);
    else if (f < 37)  build_frag(FR, f, posP.xw,  nullptr, 128, NH,  32*(f-F_WX_P), 0);
    else if (f == 37) build_frag(FR, f, posP.hw,  nullptr, NH,  NH,  0,             0);
    else if (f < 46)  build_frag(FR, f, posP.sw,  nullptr, NH,  128, 0, 16*(f-F_S_P));
    else if (f < 54)  build_frag(FR, f, posP.trw, nullptr, NH,  128, 0, 16*(f-F_T_P));
    else              build_frag(FR, f, posP.qw,  nullptr, NH,  128, 0, 16*(f-F_Q_P));
  }
  {
    const NetP* ps[2] = { &momP, &posP };
    for (int idx = tid; idx < 16; idx += WPB*32) {
      for (int net = 0; net < 2; ++net) {
        const NetP& p = *ps[net];
        const bool v = idx < NH;
        FA[O_HB0 + net*16 + idx] = v ? (p.vb[idx] + p.xb[idx] + p.tb[idx]) : 0.f;
        FA[O_WT0 + net*16 + idx] = v ? p.tw[idx]      : 0.f;
        FA[O_WT1 + net*16 + idx] = v ? p.tw[NH + idx] : 0.f;
        FA[O_BH  + net*16 + idx] = v ? p.hb[idx]      : 0.f;
      }
    }
    for (int idx = tid; idx < 128; idx += WPB*32) {
      for (int net = 0; net < 2; ++net) {
        const NetP& p = *ps[net];
        float* q = FA + O_PK + net*1024 + idx*8;
        q[0] = p.sb[idx];
        q[1] = p.trb[idx];
        q[2] = p.qb[idx];
        q[3] = __expf(p.cs[idx]);
        q[4] = __expf(p.ct[idx]);
        q[5] = 0.f; q[6] = 0.f; q[7] = 0.f;
      }
    }
  }
  half_t* tileP = WB + wave*LDS_WAVE_HALFS;
  half_t* tileA = tileP + 2048;
  half_t* tileB = tileP + 4096;
  half_t* hbuf  = tileP + 6144;
  half_t* h2buf = tileP + 6656;
  for (int j = lane; j < 512; j += 32) {     // zero pad region once
    hbuf[j]  = (half_t)0.f;
    h2buf[j] = (half_t)0.f;
  }
  __syncthreads();

  // ---- per-lane invariant biases ----
  const float hb0M = FA[O_HB0 + n],      hb0P = FA[O_HB0 + 16 + n];
  const float wt0M = FA[O_WT0 + n],      wt0P = FA[O_WT0 + 16 + n];
  const float wt1M = FA[O_WT1 + n],      wt1P = FA[O_WT1 + 16 + n];
  const float b2M  = FA[O_BH  + n],      b2P  = FA[O_BH  + 16 + n];

  // ---- load state tile into registers (WMMA C layout) ----
  const float eps  = epsPtr[0];
  const int   dir  = blockIdx.y;             // 0 = forward, 1 = backward
  const int   row0 = (blockIdx.x*WPB + wave)*16;
  const float* momSrc = dir ? noise_b : noise_f;

  float P[8][8], M[8][8], ld[8], oh[8];
#pragma unroll
  for (int i = 0; i < 8; ++i) { ld[i] = 0.f; oh[i] = 0.f; }
#pragma unroll
  for (int b = 0; b < 8; ++b)
#pragma unroll
    for (int i = 0; i < 8; ++i) {
      const int g = (row0 + i + 8*hg)*DDIM + 16*b + n;
      P[b][i] = position[g];
      M[b][i] = momSrc[g];
      oh[i] += 0.5f*(P[b][i]*P[b][i] + M[b][i]*M[b][i]);
    }

  // pre-stage the position tile for the first momentum sub-step
#pragma unroll
  for (int b = 0; b < 8; ++b)
#pragma unroll
    for (int i = 0; i < 8; ++i)
      tileP[(i + 8*hg)*DDIM + 16*b + n] = (half_t)P[b][i];
  WAVE_LDS_FENCE();

  // ---- 25-step serial scan; 4 sub-steps per step ----
#pragma unroll 1
  for (int st = 0; st < NSTEP; ++st) {
    const int sidx = dir ? (NSTEP - 1 - st) : st;
    const float ang = 6.28318530718f * (float)sidx / (float)NSTEP;
    const float t0 = __cosf(ang), t1 = __sinf(ang);
    const float* mrow = masks + sidx*DDIM;
    float m8[8];
#pragma unroll
    for (int b = 0; b < 8; ++b) m8[b] = mrow[16*b + n];
    const float hbM = fmaf(t1, wt1M, fmaf(t0, wt0M, hb0M));
    const float hbP = fmaf(t1, wt1P, fmaf(t0, wt0P, hb0P));

    if (dir == 0) {
      substep<0,1,0>(P,M,ld, tileP,tileA,tileB,hbuf,h2buf, FR,FA, m8,0, eps,hbM,b2M);
      substep<1,1,1>(P,M,ld, tileP,tileA,tileB,hbuf,h2buf, FR,FA, m8,0, eps,hbP,b2P);
      substep<1,1,0>(P,M,ld, tileP,tileA,tileB,hbuf,h2buf, FR,FA, m8,1, eps,hbP,b2P);
      substep<0,1,1>(P,M,ld, tileP,tileA,tileB,hbuf,h2buf, FR,FA, m8,0, eps,hbM,b2M);
    } else {
      substep<0,0,0>(P,M,ld, tileP,tileA,tileB,hbuf,h2buf, FR,FA, m8,0, eps,hbM,b2M);
      substep<1,0,1>(P,M,ld, tileP,tileA,tileB,hbuf,h2buf, FR,FA, m8,0, eps,hbP,b2P);
      substep<1,0,0>(P,M,ld, tileP,tileA,tileB,hbuf,h2buf, FR,FA, m8,1, eps,hbP,b2P);
      substep<0,0,1>(P,M,ld, tileP,tileA,tileB,hbuf,h2buf, FR,FA, m8,0, eps,hbM,b2M);
    }
  }

  // ---- accept prob + state writeback ----
  float* dst = ws + (size_t)dir * (size_t)(2*BD + BATCH);
#pragma unroll
  for (int i = 0; i < 8; ++i) {
    float nh = 0.f;
#pragma unroll
    for (int b = 0; b < 8; ++b) nh += 0.5f*(P[b][i]*P[b][i] + M[b][i]*M[b][i]);
    float v = oh[i] - nh + ld[i];
    v += __shfl_xor(v, 1, 16);
    v += __shfl_xor(v, 2, 16);
    v += __shfl_xor(v, 4, 16);
    v += __shfl_xor(v, 8, 16);
    if (n == 0) dst[2*BD + row0 + i + 8*hg] = __expf(fminf(v, 0.f));
  }
#pragma unroll
  for (int b = 0; b < 8; ++b)
#pragma unroll
    for (int i = 0; i < 8; ++i) {
      const int g = (row0 + i + 8*hg)*DDIM + 16*b + n;
      dst[g]      = P[b][i];
      dst[BD + g] = M[b][i];
    }
}

// ------------------------------ combine ------------------------------------

__global__ void combine_kernel(const float* __restrict__ ws,
                               const float* __restrict__ u_dir,
                               const float* __restrict__ u_acc,
                               const float* __restrict__ position,
                               float* __restrict__ out) {
  const int idx = blockIdx.x*blockDim.x + threadIdx.x;
  if (idx >= BD) return;
  const int row = idx >> 7;
  const float* pf  = ws;
  const float* mf  = ws + BD;
  const float* apf = ws + 2*BD;
  const float* pb  = ws + (2*BD + BATCH);
  const float* mb  = pb + BD;
  const float* apb = pb + 2*BD;

  const float fm = (u_dir[row] > 0.5f) ? 1.f : 0.f;
  const float bm = 1.f - fm;
  const float pp = fm*pf[idx] + bm*pb[idx];
  const float mm = fm*mf[idx] + bm*mb[idx];
  const float ap = fm*apf[row] + bm*apb[row];
  const float am = (ap > u_acc[row]) ? 1.f : 0.f;

  out[idx]           = pp;                                    // position_post
  out[BD + idx]      = mm;                                    // momentum_post
  if ((idx & 127) == 0) out[2*BD + row] = ap;                 // accept_prob
  out[2*BD + BATCH + idx] = am*pp + (1.f - am)*position[idx]; // position_out
}

// ------------------------------ launch -------------------------------------

static NetP mk_net(void* const* d, int base) {
  NetP p;
  p.cs  = (const float*)d[base + 0];   p.ct  = (const float*)d[base + 1];
  p.hw  = (const float*)d[base + 2];   p.hb  = (const float*)d[base + 3];
  p.sw  = (const float*)d[base + 4];   p.sb  = (const float*)d[base + 5];
  p.tw  = (const float*)d[base + 6];   p.tb  = (const float*)d[base + 7];
  p.qw  = (const float*)d[base + 8];   p.qb  = (const float*)d[base + 9];
  p.trw = (const float*)d[base + 10];  p.trb = (const float*)d[base + 11];
  p.vw  = (const float*)d[base + 12];  p.vb  = (const float*)d[base + 13];
  p.xw  = (const float*)d[base + 14];  p.xb  = (const float*)d[base + 15];
  return p;
}

extern "C" void kernel_launch(void* const* d_in, const int* in_sizes, int n_in,
                              void* d_out, int out_size, void* d_ws, size_t ws_size,
                              hipStream_t stream) {
  (void)in_sizes; (void)n_in; (void)out_size; (void)ws_size;
  // d_in layout (jax pytree flatten, dict keys sorted):
  // 0 position | 1..16 pos_params | 17..32 mom_params | 33 eps | 34 masks
  // 35 u_dir | 36 u_acc | 37 noise_f | 38 noise_b
  const float* position = (const float*)d_in[0];
  NetP posP = mk_net(d_in, 1);
  NetP momP = mk_net(d_in, 17);
  const float* eps     = (const float*)d_in[33];
  const float* masks   = (const float*)d_in[34];
  const float* u_dir   = (const float*)d_in[35];
  const float* u_acc   = (const float*)d_in[36];
  const float* noise_f = (const float*)d_in[37];
  const float* noise_b = (const float*)d_in[38];
  float* out = (float*)d_out;
  float* ws  = (float*)d_ws;     // needs 2*(2*BD+BATCH) floats (~8.4 MB)

  dim3 grid(BATCH/16/WPB, 2);    // 64 tiles-of-4-waves x {fwd,bwd}
  dyn_kernel<<<grid, WPB*32, LDS_BYTES, stream>>>(position, posP, momP, eps,
                                                  masks, noise_f, noise_b, ws);
  combine_kernel<<<(BD + 255)/256, 256, 0, stream>>>(ws, u_dir, u_acc,
                                                     position, out);
}
